// MeshDiT_4363686773461
// MI455X (gfx1250) — 8-bit matrix operand order repaired and checked by reading, compiled, not run on hardware
//
#include <hip/hip_runtime.h>
#include <math.h>
#include <stdint.h>


typedef __attribute__((ext_vector_type(8))) int v8i_t;

#define DEVFN static __device__ __forceinline__

DEVFN float warpRedSum(float v) {
  for (int o = 16; o > 0; o >>= 1) v += __shfl_down(v, o, 32);
  return v;
}
DEVFN float warpRedMax(float v) {
  for (int o = 16; o > 0; o >>= 1) v = fmaxf(v, __shfl_down(v, o, 32));
  return v;
}

DEVFN float blockRedSum256(float v) {
  __shared__ float sm[8];
  __shared__ float res;
  int w = threadIdx.x >> 5, l = threadIdx.x & 31;
  v = warpRedSum(v);
  __syncthreads();
  if (l == 0) sm[w] = v;
  __syncthreads();
  if (threadIdx.x == 0) {
    float s = 0.f;
    for (int i = 0; i < 8; ++i) s += sm[i];
    res = s;
  }
  __syncthreads();
  return res;
}
DEVFN float blockRedMax256(float v) {
  __shared__ float sm[8];
  __shared__ float res;
  int w = threadIdx.x >> 5, l = threadIdx.x & 31;
  v = warpRedMax(v);
  __syncthreads();
  if (l == 0) sm[w] = v;
  __syncthreads();
  if (threadIdx.x == 0) {
    float s = sm[0];
    for (int i = 1; i < 8; ++i) s = fmaxf(s, sm[i]);
    res = s;
  }
  __syncthreads();
  return res;
}

DEVFN float siluf(float x) { return x / (1.f + expf(-x)); }
DEVFN float sigmoidf_(float x) { return 1.f / (1.f + expf(-x)); }

__global__ __launch_bounds__(256) void k_wq_partial(const float* __restrict__ w,
                                                    size_t n,
                                                    float* __restrict__ partial) {
  float s = 0.f;
  for (size_t i = (size_t)blockIdx.x * 256 + threadIdx.x; i < n;
       i += (size_t)gridDim.x * 256)
    s += fabsf(w[i]);
  s = blockRedSum256(s);
  if (threadIdx.x == 0) partial[blockIdx.x] = s;
}

__global__ __launch_bounds__(256) void k_wq_final(const float* __restrict__ partial,
                                                  size_t n,
                                                  float* __restrict__ sc) {
  float v = partial[threadIdx.x];
  v = blockRedSum256(v);
  if (threadIdx.x == 0) {
    float mean = v / (float)n;
    float inv = fmaxf(mean, 1e-5f);
    sc[0] = 1.f / inv;
    sc[1] = inv;
  }
}

__global__ __launch_bounds__(256) void k_wq_apply(const float* __restrict__ w,
                                                  int8_t* __restrict__ q, size_t n,
                                                  const float* __restrict__ sc) {
  float s = sc[0];
  for (size_t i = (size_t)blockIdx.x * 256 + threadIdx.x; i < n;
       i += (size_t)gridDim.x * 256) {
    float v = rintf(w[i] * s);
    v = fminf(fmaxf(v, -1.f), 1.f);
    q[i] = (int8_t)v;
  }
}

__global__ __launch_bounds__(256) void k_prep(const float* __restrict__ x,
                                              const float* __restrict__ nrm,
                                              int8_t* __restrict__ q,
                                              float* __restrict__ arow, int K) {
  int row = blockIdx.x;
  const float* xr = x + (size_t)row * K;
  float cache[18];
  int cnt = 0;
  float ss = 0.f;
  for (int i = threadIdx.x; i < K; i += 256) {
    float v = xr[i];
    cache[cnt++] = v;
    ss += v * v;
  }
  float ms = blockRedSum256(ss) / (float)K;
  float rr = rsqrtf(ms + 1e-6f);
  float amax = 0.f;
  cnt = 0;
  for (int i = threadIdx.x; i < K; i += 256) {
    float v = cache[cnt] * rr * nrm[i];
    cache[cnt++] = v;
    amax = fmaxf(amax, fabsf(v));
  }
  amax = blockRedMax256(amax);
  float s = 127.f / fmaxf(amax, 1e-5f);
  if (threadIdx.x == 0) arow[row] = 1.f / s;
  int8_t* qr = q + (size_t)row * K;
  cnt = 0;
  for (int i = threadIdx.x; i < K; i += 256) {
    float v = rintf(cache[cnt++] * s);
    v = fminf(fmaxf(v, -128.f), 127.f);
    qr[i] = (int8_t)v;
  }
}

template <int MT>
__global__ __launch_bounds__(256) void k_gemm_i8(
    const int8_t* __restrict__ Aq, const int8_t* __restrict__ Wq,
    const float* __restrict__ arow, const float* __restrict__ wsc,
    const float* __restrict__ bias, float* __restrict__ out, int M, int N, int K) {
  int wave = (int)((blockIdx.x * 256u + threadIdx.x) >> 5);
  int lane = threadIdx.x & 31;
  int tilesN = N >> 6;
  int tilesM = M >> (MT == 2 ? 5 : 4);
  if (wave >= tilesM * tilesN) return;
  int tm = wave / tilesN;
  int tn = wave - tm * tilesN;
  int m0 = tm << (MT == 2 ? 5 : 4);
  int n0 = tn << 6;
  int half = lane >> 4;
  int l15 = lane & 15;

  v8i_t acc[MT][4];
#pragma unroll
  for (int i = 0; i < MT; ++i)
#pragma unroll
    for (int j = 0; j < 4; ++j) acc[i][j] = (v8i_t){};

  const int8_t* ap = Aq + (size_t)(m0 + l15) * K + (half << 3);
  const int8_t* bp = Wq + (size_t)(n0 + l15) * K + (half << 3);
  const size_t mstride = (size_t)16 * K;

  auto loadA = [&](int i, int k0) -> v8i_t {
    const uint2* p = (const uint2*)(ap + (size_t)i * mstride + k0);
    uint2 q0 = p[0], q1 = p[2], q2 = p[4], q3 = p[6];
    v8i_t a;
    a[0] = (int)q0.x; a[1] = (int)q0.y;
    a[2] = (int)q1.x; a[3] = (int)q1.y;
    a[4] = (int)q2.x; a[5] = (int)q2.y;
    a[6] = (int)q3.x; a[7] = (int)q3.y;
    return a;
  };
  auto loadB = [&](int j, int k0) -> v8i_t {
    const uint2* p = (const uint2*)(bp + (size_t)j * mstride + k0);
    uint2 q0 = p[0], q1 = p[2], q2 = p[4], q3 = p[6];
    v8i_t b;
    b[0] = (int)q0.x; b[1] = (int)q0.y; b[2] = (int)q1.x; b[3] = (int)q1.y;
    b[4] = (int)q2.x; b[5] = (int)q2.y; b[6] = (int)q3.x; b[7] = (int)q3.y;
    return b;
  };
  auto loadAll = [&](v8i_t (&a)[MT], v8i_t (&b)[4], int k0) {
    __builtin_prefetch(ap + k0 + 128, 0, 3);
    __builtin_prefetch(bp + k0 + 128, 0, 3);
#pragma unroll
    for (int i = 0; i < MT; ++i) a[i] = loadA(i, k0);
#pragma unroll
    for (int j = 0; j < 4; ++j) b[j] = loadB(j, k0);
  };
  auto mmaAll = [&](v8i_t (&a)[MT], v8i_t (&b)[4]) {
#pragma unroll
    for (int j = 0; j < 4; ++j)
#pragma unroll
      for (int i = 0; i < MT; ++i)
        acc[i][j] = __builtin_amdgcn_wmma_i32_16x16x64_iu8(true, a[i], true, b[j],
                                                           acc[i][j], false, false);
  };

  v8i_t aP[MT], bP[4], aQ[MT], bQ[4];
  const int nsteps = K >> 6;
  loadAll(aP, bP, 0);
  int s = 0;
  while (s + 1 < nsteps) {
    loadAll(aQ, bQ, (s + 1) << 6);
    mmaAll(aP, bP);
    ++s;
    if (s + 1 < nsteps) loadAll(aP, bP, (s + 1) << 6);
    mmaAll(aQ, bQ);
    ++s;
  }
  if (s < nsteps) mmaAll(aP, bP);

  float wsv = wsc[0];
#pragma unroll
  for (int i = 0; i < MT; ++i) {
#pragma unroll
    for (int j = 0; j < 4; ++j) {
      int n = n0 + j * 16 + l15;
      float bv = bias ? bias[n] : 0.f;
#pragma unroll
      for (int r = 0; r < 8; ++r) {
        int m = m0 + i * 16 + r + half * 8;
        out[(size_t)m * N + n] = (float)acc[i][j][r] * arow[m] * wsv + bv;
      }
    }
  }
}

__global__ __launch_bounds__(128) void k_linear(const float* __restrict__ X, int ldx,
                                                const float* __restrict__ W,
                                                const float* __restrict__ bias,
                                                float* __restrict__ out, int ldo,
                                                int K) {
  int n = blockIdx.x, m = blockIdx.y;
  const float* xr = X + (size_t)m * ldx;
  const float* wr = W + (size_t)n * K;
  float s = 0.f;
  for (int i = threadIdx.x; i < K; i += 128) s += xr[i] * wr[i];
  s = warpRedSum(s);
  __shared__ float sm[4];
  int w = threadIdx.x >> 5, l = threadIdx.x & 31;
  if (l == 0) sm[w] = s;
  __syncthreads();
  if (threadIdx.x == 0)
    out[(size_t)m * ldo + n] = sm[0] + sm[1] + sm[2] + sm[3] + (bias ? bias[n] : 0.f);
}

__global__ void k_silu(const float* __restrict__ x, float* __restrict__ y, int n) {
  int i = blockIdx.x * 256 + threadIdx.x;
  if (i < n) y[i] = siluf(x[i]);
}

__global__ void k_silu_mul(const float* __restrict__ src, float* __restrict__ dst,
                           long total, int F) {
  long i = (long)blockIdx.x * 256 + threadIdx.x;
  if (i >= total) return;
  long m = i / F;
  int j = (int)(i - m * F);
  const float* srow = src + m * (2L * F);
  dst[i] = siluf(srow[j]) * srow[F + j];
}

__global__ void k_add3(const float* a, const float* b, const float* c,
                       float* __restrict__ o, int n) {
  int i = blockIdx.x * 256 + threadIdx.x;
  if (i < n) o[i] = a[i] + b[i] + c[i];
}

__global__ void k_add_pos(float* __restrict__ h, const float* __restrict__ pos,
                          long n, int H, int L) {
  long i = (long)blockIdx.x * 256 + threadIdx.x;
  if (i >= n) return;
  long row = i / H;
  int col = (int)(i - row * H);
  int l = (int)(row % L);
  h[i] += pos[(size_t)l * H + col];
}

__global__ void k_timestep(const float* __restrict__ t, float* __restrict__ out) {
  int b = blockIdx.x, j = threadIdx.x;
  float tv = t[b];
  float v;
  if (j < 128) {
    float f = expf(-logf(10000.f) * (float)j / 128.f);
    v = cosf(tv * f);
  } else {
    float f = expf(-logf(10000.f) * (float)(j - 128) / 128.f);
    v = sinf(tv * f);
  }
  out[b * 256 + j] = v;
}

__global__ void k_text_pool(const int* __restrict__ ids, const int* __restrict__ mask,
                            const float* __restrict__ table, float* __restrict__ out,
                            int H) {
  int h = blockIdx.x * 256 + threadIdx.x;
  int b = blockIdx.y;
  if (h >= H) return;
  float acc = 0.f, msum = 0.f;
  for (int tt = 0; tt < 77; ++tt) {
    float mk = (float)mask[b * 77 + tt];
    acc += mk * table[(size_t)ids[b * 77 + tt] * H + h];
    msum += mk;
  }
  out[(size_t)b * H + h] = acc / fmaxf(msum, 1e-9f);
}

__global__ __launch_bounds__(256) void k_ln_mod(const float* __restrict__ x,
                                                const float* __restrict__ mods,
                                                int mstride, int sOff, int scOff,
                                                float* __restrict__ out, int Lseq,
                                                int Hd) {
  int row = blockIdx.x;
  int b = row / Lseq;
  const float* xr = x + (size_t)row * Hd;
  float s = 0.f, ss = 0.f;
  for (int i = threadIdx.x; i < Hd; i += 256) {
    float v = xr[i];
    s += v;
    ss += v * v;
  }
  s = blockRedSum256(s);
  ss = blockRedSum256(ss);
  float mean = s / (float)Hd;
  float var = ss / (float)Hd - mean * mean;
  float rr = rsqrtf(var + 1e-6f);
  const float* sh = mods + (size_t)b * mstride + sOff;
  const float* sc = mods + (size_t)b * mstride + scOff;
  float* orow = out + (size_t)row * Hd;
  for (int i = threadIdx.x; i < Hd; i += 256)
    orow[i] = (xr[i] - mean) * rr * (1.f + sc[i]) + sh[i];
}

__global__ void k_sig_silu(float* __restrict__ ib, float* __restrict__ fb, long n) {
  long i = (long)blockIdx.x * 256 + threadIdx.x;
  if (i >= n) return;
  float f = sigmoidf_(fb[i]);
  float ii = siluf(ib[i]) * (1.f - f);
  fb[i] = f;
  ib[i] = ii;
}

__global__ void k_scan(const float* __restrict__ f, const float* __restrict__ ii,
                       float* __restrict__ o, int Bn, int Lseq, int Hd) {
  int c = blockIdx.x * 256 + threadIdx.x;
  if (c >= Bn * Hd) return;
  int b = c / Hd;
  int h = c - b * Hd;
  size_t base = (size_t)b * Lseq * Hd + h;
  float hp = 0.f;
  for (int t = 0; t < Lseq; ++t) {
    size_t idx = base + (size_t)t * Hd;
    hp = f[idx] * hp + ii[idx];
    o[idx] = hp;
  }
}

__global__ __launch_bounds__(256) void k_rms_gate(float* __restrict__ o,
                                                  const float* __restrict__ g,
                                                  const float* __restrict__ gn,
                                                  int Hd) {
  int row = blockIdx.x;
  float* orow = o + (size_t)row * Hd;
  const float* grow = g + (size_t)row * Hd;
  float ss = 0.f;
  for (int i = threadIdx.x; i < Hd; i += 256) {
    float v = orow[i];
    ss += v * v;
  }
  ss = blockRedSum256(ss);
  float rr = rsqrtf(ss / (float)Hd + 1e-6f);
  for (int i = threadIdx.x; i < Hd; i += 256)
    orow[i] = orow[i] * rr * gn[i] * siluf(grow[i]);
}

__global__ void k_resid(float* __restrict__ h, const float* __restrict__ y,
                        const float* __restrict__ mods, int mstride, int gOff,
                        int Lseq, int Hd, long n) {
  long i = (long)blockIdx.x * 256 + threadIdx.x;
  if (i >= n) return;
  long row = i / Hd;
  int col = (int)(i - row * Hd);
  int b = (int)(row / Lseq);
  h[i] += mods[(size_t)b * mstride + gOff + col] * y[i];
}

__global__ __launch_bounds__(256) void k_rms_resid(float* __restrict__ h,
                                                   const float* __restrict__ m,
                                                   const float* __restrict__ w,
                                                   const float* __restrict__ mods,
                                                   int mstride, int gOff, int Lseq,
                                                   int Hd) {
  int row = blockIdx.x;
  int b = row / Lseq;
  const float* mrow = m + (size_t)row * Hd;
  float* hrow = h + (size_t)row * Hd;
  const float* gm = mods + (size_t)b * mstride + gOff;
  float ss = 0.f;
  for (int i = threadIdx.x; i < Hd; i += 256) {
    float v = mrow[i];
    ss += v * v;
  }
  ss = blockRedSum256(ss);
  float rr = rsqrtf(ss / (float)Hd + 1e-6f);
  for (int i = threadIdx.x; i < Hd; i += 256)
    hrow[i] += gm[i] * (mrow[i] * rr * w[i]);
}

__global__ __launch_bounds__(256) void k_rms_w(const float* __restrict__ x,
                                               const float* __restrict__ w,
                                               float* __restrict__ out, int D) {
  int row = blockIdx.x;
  const float* xr = x + (size_t)row * D;
  float ss = 0.f;
  for (int i = threadIdx.x; i < D; i += 256) {
    float v = xr[i];
    ss += v * v;
  }
  ss = blockRedSum256(ss);
  float rr = rsqrtf(ss / (float)D + 1e-6f);
  float* orow = out + (size_t)row * D;
  for (int i = threadIdx.x; i < D; i += 256) orow[i] = xr[i] * rr * w[i];
}

extern "C" void kernel_launch(void* const* d_in, const int* in_sizes, int n_in,
                              void* d_out, int out_size, void* d_ws, size_t ws_size,
                              hipStream_t stream) {
  (void)in_sizes; (void)n_in; (void)out_size; (void)ws_size;
  const int H = 1152, L = 2048, Bn = 4, M = Bn * L;
  auto F32 = [&](int i) { return (const float*)d_in[i]; };
  auto I32 = [&](int i) { return (const int*)d_in[i]; };

  char* ws = (char*)d_ws;
  size_t cur = 0;
  auto alloc = [&](size_t bytes) -> char* {
    char* p = ws + cur;
    cur = (cur + bytes + 255) & ~(size_t)255;
    return p;
  };

  const int NW = 22;
  const int bw_wi[NW] = {5, 14, 16,
                         22, 24, 26, 29, 31, 33, 37, 39,
                         46, 48, 50, 53, 55, 57, 61, 63,
                         71, 73, 80};
  const int bw_N[NW] = {1152, 6144, 1152,
                        1152, 1152, 1152, 1152, 9216, 1152, 6144, 1152,
                        1152, 1152, 1152, 1152, 9216, 1152, 6144, 1152,
                        6144, 1152, 64};
  const int bw_K[NW] = {64, 1152, 3072,
                        1152, 1152, 1152, 1152, 1152, 4608, 1152, 3072,
                        1152, 1152, 1152, 1152, 1152, 4608, 1152, 3072,
                        1152, 3072, 1152};

  int8_t* wq[NW];
  for (int t = 0; t < NW; ++t) wq[t] = (int8_t*)alloc((size_t)bw_N[t] * bw_K[t]);
  float* wscales = (float*)alloc(NW * 2 * sizeof(float));
  float* part = (float*)alloc(256 * sizeof(float));

  int8_t* AQ = (int8_t*)alloc((size_t)M * 4608);
  float* AROW = (float*)alloc((size_t)M * sizeof(float));

  float* BH = (float*)alloc((size_t)M * H * 4);
  float* BX = (float*)alloc((size_t)M * H * 4);
  float* BI = (float*)alloc((size_t)M * H * 4);
  float* BF = (float*)alloc((size_t)M * H * 4);
  float* BG = (float*)alloc((size_t)M * H * 4);
  float* BO = (float*)alloc((size_t)M * H * 4);
  float* BGATE = (float*)alloc((size_t)M * 9216 * 4);
  float* BMID = (float*)alloc((size_t)M * 4608 * 4);

  size_t condStart = cur;
  float* CT = (float*)alloc(16 * 256 * 4);
  float* CA = (float*)alloc(16 * 1152 * 4);
  float* CB = (float*)alloc(16 * 1152 * 4);
  float* CPOOL = (float*)alloc(16 * 1152 * 4);
  float* CC = (float*)alloc(16 * 1152 * 4);
  float* CCS = (float*)alloc(16 * 1152 * 4);
  float* CG = (float*)alloc(16 * 6144 * 4);
  float* CM = (float*)alloc(16 * 3072 * 4);
  float* CD = (float*)alloc(16 * 1152 * 4);
  float* CO1 = (float*)alloc(16 * 6912 * 4);
  float* CO2 = (float*)alloc(16 * 6912 * 4);
  float* CMODS = (float*)alloc(16 * 6912 * 4);
  size_t condBytes = cur - condStart;
  hipMemsetAsync(ws + condStart, 0, condBytes, stream);

  for (int t = 0; t < NW; ++t) {
    size_t n = (size_t)bw_N[t] * bw_K[t];
    k_wq_partial<<<256, 256, 0, stream>>>(F32(bw_wi[t]), n, part);
    k_wq_final<<<1, 256, 0, stream>>>(part, n, wscales + t * 2);
    int blocks = (int)((n + 2047) / 2048);
    if (blocks > 8192) blocks = 8192;
    k_wq_apply<<<blocks, 256, 0, stream>>>(F32(bw_wi[t]), wq[t], n, wscales + t * 2);
  }

  auto bitlin = [&](const float* x, int rows, int nrmIdx, int t, const float* bias,
                    float* out) {
    k_prep<<<rows, 256, 0, stream>>>(x, F32(nrmIdx), AQ, AROW, bw_K[t]);
    if ((rows & 31) == 0) {
      int waves = (rows >> 5) * (bw_N[t] >> 6);
      k_gemm_i8<2><<<(waves * 32 + 255) / 256, 256, 0, stream>>>(
          AQ, wq[t], AROW, wscales + t * 2 + 1, bias, out, rows, bw_N[t], bw_K[t]);
    } else {
      int waves = (rows >> 4) * (bw_N[t] >> 6);
      k_gemm_i8<1><<<(waves * 32 + 255) / 256, 256, 0, stream>>>(
          AQ, wq[t], AROW, wscales + t * 2 + 1, bias, out, rows, bw_N[t], bw_K[t]);
    }
  };
  auto dense = [&](const float* X, int ldx, int wIdx, int N, int K, const float* bias,
                   float* out, int ldo, int rows) {
    dim3 g(N, rows);
    k_linear<<<g, 128, 0, stream>>>(X, ldx, F32(wIdx), bias, out, ldo, K);
  };

  bitlin(F32(0), M, 6, 0, F32(7), BH);
  k_add_pos<<<(int)(((long)M * H + 255) / 256), 256, 0, stream>>>(BH, F32(8),
                                                                  (long)M * H, H, L);

  k_timestep<<<4, 256, 0, stream>>>(F32(1), CT);
  dense(CT, 256, 9, H, 256, F32(10), CA, H, 4);
  k_silu<<<(4 * H + 255) / 256, 256, 0, stream>>>(CA, CA, 4 * H);
  dense(CA, H, 11, H, H, F32(12), CB, H, 4);

  k_text_pool<<<dim3((H + 255) / 256, 4), 256, 0, stream>>>(I32(3), I32(4), F32(13),
                                                            CPOOL, H);
  bitlin(CPOOL, 16, 15, 1, nullptr, CG);
  k_silu_mul<<<(16 * 3072 + 255) / 256, 256, 0, stream>>>(CG, CM, 16L * 3072, 3072);
  bitlin(CM, 16, 17, 2, nullptr, CD);

  dense(F32(2), 16384, 18, H, 16384, F32(19), CA, H, 4);
  k_silu<<<(4 * H + 255) / 256, 256, 0, stream>>>(CA, CA, 4 * H);
  dense(CA, H, 20, H, H, F32(21), CPOOL, H, 4);

  k_add3<<<(16 * H + 255) / 256, 256, 0, stream>>>(CB, CD, CPOOL, CC, 16 * H);
  k_silu<<<(16 * H + 255) / 256, 256, 0, stream>>>(CC, CCS, 16 * H);

  auto adaln = [&](int inW, int agT, int agNorm, int adT, int adNorm, int outpW,
                   int outpB, int normIdx, int outW, int outB, int outDim,
                   float* mods) {
    dense(CCS, H, inW, H, H, nullptr, CA, H, 4);
    bitlin(CA, 16, agNorm, agT, nullptr, CG);
    k_silu_mul<<<(16 * 3072 + 255) / 256, 256, 0, stream>>>(CG, CM, 16L * 3072, 3072);
    bitlin(CM, 16, adNorm, adT, nullptr, CD);
    dense(CD, H, outpW, outDim, H, F32(outpB), CO1, outDim, 4);
    k_rms_w<<<4, 256, 0, stream>>>(CO1, F32(normIdx), CO2, outDim);
    dense(CO2, outDim, outW, outDim, outDim, F32(outB), mods, outDim, 4);
  };

  const long nMH = (long)M * H;
  const int eBlocks = (int)((nMH + 255) / 256);

  for (int blk = 0; blk < 2; ++blk) {
    int base = 22 + 24 * blk;
    int ti = 3 + 8 * blk;

    adaln(base + 14, ti + 6, base + 16, ti + 7, base + 18, base + 19, base + 20,
          base + 21, base + 22, base + 23, 6912, CMODS);

    k_ln_mod<<<M, 256, 0, stream>>>(BH, CMODS, 6912, 0, 1152, BX, L, H);
    bitlin(BX, M, base + 1, ti + 0, nullptr, BI);
    bitlin(BX, M, base + 3, ti + 1, nullptr, BF);
    k_sig_silu<<<eBlocks, 256, 0, stream>>>(BI, BF, nMH);
    bitlin(BX, M, base + 5, ti + 2, nullptr, BG);
    k_scan<<<(Bn * H + 255) / 256, 256, 0, stream>>>(BF, BI, BO, Bn, L, H);
    k_rms_gate<<<M, 256, 0, stream>>>(BO, BG, F32(base + 6), H);
    bitlin(BO, M, base + 8, ti + 3, nullptr, BX);
    k_resid<<<eBlocks, 256, 0, stream>>>(BH, BX, CMODS, 6912, 2304, L, H, nMH);

    k_ln_mod<<<M, 256, 0, stream>>>(BH, CMODS, 6912, 3456, 4608, BX, L, H);
    bitlin(BX, M, base + 10, ti + 4, nullptr, BGATE);
    {
      long tot = (long)M * 4608;
      k_silu_mul<<<(int)((tot + 255) / 256), 256, 0, stream>>>(BGATE, BMID, tot, 4608);
    }
    bitlin(BMID, M, base + 12, ti + 5, nullptr, BX);
    k_rms_resid<<<M, 256, 0, stream>>>(BH, BX, F32(base + 13), CMODS, 6912, 5760, L, H);
  }

  adaln(70, 19, 72, 20, 74, 75, 76, 77, 78, 79, 2304, CMODS);
  k_ln_mod<<<M, 256, 0, stream>>>(BH, CMODS, 2304, 0, 1152, BX, L, H);
  bitlin(BX, M, 81, 21, F32(82), (float*)d_out);
}
